// HelmBlock_31009663877801
// MI455X (gfx1250) — matrix operand order repaired and checked by reading, not compiled, not run on hardware
//
#include <hip/hip_runtime.h>
#include <math.h>

typedef __attribute__((ext_vector_type(16))) _Float16 v16h;
typedef __attribute__((ext_vector_type(8)))  float    v8f;

#define NB   8        // B*G batch groups
#define CG   8        // channels per group
#define S0   66
#define S0C  287496   // 66^3
#define S1   33
#define S1C  35937    // 33^3
#define TDC  262144   // 64^3
#define EPSV 1e-5f

// ---- output offsets (in floats), concatenated return order ----
#define X_OFF     0
#define HELM_OFF  16777216   // (8,4,66^3)
#define VEL_OFF   25977088   // (8,3,64^3)
#define VELP_OFF  32268544
#define VELV_OFF  38560000

// ======================= encoder: grouped 2x2x2 stride-2 conv ==============
// in: full (64, 66^3) tensor viewed as (8 groups, 8 ch). msk broadcast. out (8,64,33^3)
__global__ void enc_kernel(const float* __restrict__ in, const float* __restrict__ msk,
                           const float* __restrict__ w, const float* __restrict__ bias,
                           float* __restrict__ out) {
  __shared__ float wl[64 * 64];   // (o, c*8+tap)
  __shared__ float bl[64];
  for (int i = threadIdx.x; i < 4096; i += blockDim.x) wl[i] = w[i];
  if (threadIdx.x < 64) bl[threadIdx.x] = bias[threadIdx.x];
  __syncthreads();
  int t = blockIdx.x * blockDim.x + threadIdx.x;
  if (t >= NB * S1C) return;
  int b = t / S1C, vox = t % S1C;
  int x = vox / (S1 * S1), y = (vox / S1) % S1, z = vox % S1;
  float acc[64];
#pragma unroll
  for (int o = 0; o < 64; ++o) acc[o] = bl[o];
  for (int c = 0; c < CG; ++c) {
    const float* ip = in + (size_t)(b * CG + c) * S0C;
#pragma unroll
    for (int tap = 0; tap < 8; ++tap) {
      int dx = tap >> 2, dy = (tap >> 1) & 1, dz = tap & 1;
      int idx = ((2 * x + dx) * S0 + (2 * y + dy)) * S0 + (2 * z + dz);
      float v = ip[idx] * msk[idx];
      const float* wp = &wl[c * 8 + tap];
#pragma unroll
      for (int o = 0; o < 64; ++o) acc[o] = fmaf(v, wp[o * 64], acc[o]);
    }
  }
  float* op = out + (size_t)b * 64 * S1C + vox;
#pragma unroll
  for (int o = 0; o < 64; ++o) op[(size_t)o * S1C] = acc[o];
}

// ======================= 3D correlation (27 channels per call) =============
__global__ void corr_kernel(const float* __restrict__ fp, const float* __restrict__ fn,
                            float* __restrict__ corr, int cofs) {
  int t = blockIdx.x * blockDim.x + threadIdx.x;
  if (t >= NB * S1C) return;
  int b = t / S1C, vox = t % S1C;
  int x = vox / (S1 * S1), y = (vox / S1) % S1, z = vox % S1;
  int offs[9]; float okm[9];
  int s = 0;
  for (int di = -1; di <= 1; ++di)
    for (int dj = -1; dj <= 1; ++dj, ++s) {
      int xx = x + di, yy = y + dj;
      bool ok = (xx >= 0 && xx < S1 && yy >= 0 && yy < S1);
      okm[s] = ok ? 1.f : 0.f;
      offs[s] = ok ? ((xx * S1 + yy) * S1 + z) : 0;
    }
  float dot[9];
#pragma unroll
  for (int i = 0; i < 9; ++i) dot[i] = 0.f;
  const float* fpb = fp + (size_t)b * 64 * S1C + vox;
  const float* fnb = fn + (size_t)b * 64 * S1C;
  for (int c = 0; c < 64; ++c) {
    float p = fpb[(size_t)c * S1C];
    const float* fc = fnb + (size_t)c * S1C;
#pragma unroll
    for (int i = 0; i < 9; ++i) dot[i] = fmaf(p, fc[offs[i]] * okm[i], dot[i]);
  }
  float zm0 = (z == S1 - 1) ? 0.f : 1.f;
  float zm2 = (z == 0) ? 0.f : 1.f;
  const float inv = 1.0f / 64.0f;
  float* cb = corr + (size_t)b * 54 * S1C + (size_t)cofs * S1C + vox;
#pragma unroll
  for (int i = 0; i < 9; ++i) {
    float dv = dot[i] * inv;
    cb[(size_t)i * S1C]        = dv * zm0;
    cb[(size_t)(9 + i) * S1C]  = dv;
    cb[(size_t)(18 + i) * S1C] = dv * zm2;
  }
}

// ======================= conv1: 54->8, 3x3x3, pad 1, 33^3 ==================
__global__ void conv1_kernel(const float* __restrict__ corr, const float* __restrict__ w,
                             float* __restrict__ out) {
  __shared__ float wl[8 * 54 * 27];  // 46656 B
  for (int i = threadIdx.x; i < 8 * 54 * 27; i += blockDim.x) wl[i] = w[i];
  __syncthreads();
  int t = blockIdx.x * blockDim.x + threadIdx.x;
  if (t >= NB * S1C) return;
  int b = t / S1C, vox = t % S1C;
  int x = vox / (S1 * S1), y = (vox / S1) % S1, z = vox % S1;
  int offs[27]; float okm[27];
  int s = 0;
  for (int dx = -1; dx <= 1; ++dx)
    for (int dy = -1; dy <= 1; ++dy)
      for (int dz = -1; dz <= 1; ++dz, ++s) {
        int xx = x + dx, yy = y + dy, zz = z + dz;
        bool ok = (xx >= 0 && xx < S1 && yy >= 0 && yy < S1 && zz >= 0 && zz < S1);
        okm[s] = ok ? 1.f : 0.f;
        offs[s] = ok ? ((xx * S1 + yy) * S1 + zz) : 0;
      }
  float acc[8];
#pragma unroll
  for (int o = 0; o < 8; ++o) acc[o] = 0.f;
  for (int ci = 0; ci < 54; ++ci) {
    const float* ip = corr + ((size_t)b * 54 + ci) * S1C;
    for (int tp = 0; tp < 27; ++tp) {
      float v = ip[offs[tp]] * okm[tp];
      const float* wp = &wl[ci * 27 + tp];
#pragma unroll
      for (int o = 0; o < 8; ++o) acc[o] = fmaf(v, wp[o * 1458], acc[o]);
    }
  }
#pragma unroll
  for (int o = 0; o < 8; ++o) out[((size_t)b * 8 + o) * S1C + vox] = acc[o];
}

// ======================= BN stats: deterministic 2-stage reduce ============
__global__ void bnstat1_kernel(const float* __restrict__ src, float* __restrict__ part,
                               int nper, int nch, int nb) {
  int ch = blockIdx.y;
  int total = nb * nper;
  float s = 0.f, q = 0.f;
  for (int e = blockIdx.x * blockDim.x + threadIdx.x; e < total; e += gridDim.x * blockDim.x) {
    int b = e / nper, vox = e - b * nper;
    float v = src[((size_t)(b * nch + ch)) * nper + vox];
    s += v; q += v * v;
  }
  __shared__ float rs[256], rq[256];
  rs[threadIdx.x] = s; rq[threadIdx.x] = q;
  __syncthreads();
  for (int o = 128; o > 0; o >>= 1) {
    if (threadIdx.x < o) { rs[threadIdx.x] += rs[threadIdx.x + o]; rq[threadIdx.x] += rq[threadIdx.x + o]; }
    __syncthreads();
  }
  if (threadIdx.x == 0) {
    part[(ch * gridDim.x + blockIdx.x) * 2]     = rs[0];
    part[(ch * gridDim.x + blockIdx.x) * 2 + 1] = rq[0];
  }
}

__global__ void bnstat2_kernel(const float* __restrict__ part, float* __restrict__ stats, int nblk) {
  int ch = blockIdx.x;
  __shared__ float rs[64], rq[64];
  float s = 0.f, q = 0.f;
  for (int i = threadIdx.x; i < nblk; i += blockDim.x) {
    s += part[(ch * nblk + i) * 2];
    q += part[(ch * nblk + i) * 2 + 1];
  }
  rs[threadIdx.x] = s; rq[threadIdx.x] = q;
  __syncthreads();
  for (int o = 32; o > 0; o >>= 1) {
    if (threadIdx.x < o) { rs[threadIdx.x] += rs[threadIdx.x + o]; rq[threadIdx.x] += rq[threadIdx.x + o]; }
    __syncthreads();
  }
  if (threadIdx.x == 0) { stats[ch * 2] = rs[0]; stats[ch * 2 + 1] = rq[0]; }
}

// ========== BN1+ReLU fused into 2x trilinear (align-corners-like) upsample =
__global__ void upsample_kernel(const float* __restrict__ src, const float* __restrict__ stats,
                                const float* __restrict__ g, const float* __restrict__ bb,
                                float invc, float* __restrict__ dst) {
  int t = blockIdx.x * blockDim.x + threadIdx.x;
  if (t >= NB * S0C) return;
  int b = t / S0C, ov = t % S0C;
  int x = ov / (S0 * S0), y = (ov / S0) % S0, z = ov % S0;
  const float sc = 32.0f / 65.0f;
  float fx = x * sc, fy = y * sc, fz = z * sc;
  int x0 = (int)fx, y0 = (int)fy, z0 = (int)fz;
  int x1 = min(x0 + 1, 32), y1 = min(y0 + 1, 32), z1 = min(z0 + 1, 32);
  float wx = fx - x0, wy = fy - y0, wz = fz - z0;
  for (int c = 0; c < 8; ++c) {
    float mean = stats[2 * c] * invc;
    float var  = stats[2 * c + 1] * invc - mean * mean;
    float rsd  = rsqrtf(var + EPSV);
    float a = rsd * g[c], sh = bb[c] - mean * a;
    const float* sp = src + ((size_t)b * 8 + c) * S1C;
#define RD(xi, yi, zi) fmaxf(sp[((xi) * S1 + (yi)) * S1 + (zi)] * a + sh, 0.f)
    float v00 = RD(x0, y0, z0) * (1.f - wz) + RD(x0, y0, z1) * wz;
    float v01 = RD(x0, y1, z0) * (1.f - wz) + RD(x0, y1, z1) * wz;
    float v10 = RD(x1, y0, z0) * (1.f - wz) + RD(x1, y0, z1) * wz;
    float v11 = RD(x1, y1, z0) * (1.f - wz) + RD(x1, y1, z1) * wz;
#undef RD
    float v = (v00 * (1.f - wy) + v01 * wy) * (1.f - wx) + (v10 * (1.f - wy) + v11 * wy) * wx;
    dst[((size_t)b * 8 + c) * S0C + ov] = v;
  }
}

// ======================= conv2: 8->8, 3x3x3, pad 1, 66^3 ===================
__global__ void conv2_kernel(const float* __restrict__ up, const float* __restrict__ w,
                             float* __restrict__ out) {
  __shared__ float wl[8 * 8 * 27];  // 6912 B
  for (int i = threadIdx.x; i < 8 * 8 * 27; i += blockDim.x) wl[i] = w[i];
  __syncthreads();
  int t = blockIdx.x * blockDim.x + threadIdx.x;
  if (t >= NB * S0C) return;
  int b = t / S0C, vox = t % S0C;
  int x = vox / (S0 * S0), y = (vox / S0) % S0, z = vox % S0;
  int offs[27]; float okm[27];
  int s = 0;
  for (int dx = -1; dx <= 1; ++dx)
    for (int dy = -1; dy <= 1; ++dy)
      for (int dz = -1; dz <= 1; ++dz, ++s) {
        int xx = x + dx, yy = y + dy, zz = z + dz;
        bool ok = (xx >= 0 && xx < S0 && yy >= 0 && yy < S0 && zz >= 0 && zz < S0);
        okm[s] = ok ? 1.f : 0.f;
        offs[s] = ok ? ((xx * S0 + yy) * S0 + zz) : 0;
      }
  float acc[8];
#pragma unroll
  for (int o = 0; o < 8; ++o) acc[o] = 0.f;
  for (int ci = 0; ci < 8; ++ci) {
    const float* ip = up + ((size_t)b * 8 + ci) * S0C;
    for (int tp = 0; tp < 27; ++tp) {
      float v = ip[offs[tp]] * okm[tp];
      const float* wp = &wl[ci * 27 + tp];
#pragma unroll
      for (int o = 0; o < 8; ++o) acc[o] = fmaf(v, wp[o * 216], acc[o]);
    }
  }
#pragma unroll
  for (int o = 0; o < 8; ++o) out[((size_t)b * 8 + o) * S0C + vox] = acc[o];
}

// ======== BN2+ReLU + 1x1 conv + scalar weight → helmholtz channels =========
__global__ void finalize_kernel(const float* __restrict__ c2, const float* __restrict__ stats,
                                const float* __restrict__ g, const float* __restrict__ bb,
                                const float* __restrict__ ow, const float* __restrict__ ob,
                                const float* __restrict__ wsc, float wmul, float invc,
                                float* __restrict__ helm, int chbase, int nout) {
  int t = blockIdx.x * blockDim.x + threadIdx.x;
  if (t >= NB * S0C) return;
  int b = t / S0C, ov = t % S0C;
  float h[8];
#pragma unroll
  for (int c = 0; c < 8; ++c) {
    float mean = stats[2 * c] * invc;
    float var  = stats[2 * c + 1] * invc - mean * mean;
    float rsd  = rsqrtf(var + EPSV);
    float a = rsd * g[c], sh = bb[c] - mean * a;
    h[c] = fmaxf(c2[((size_t)b * 8 + c) * S0C + ov] * a + sh, 0.f);
  }
  float sfac = wsc[0] * wmul;
  for (int o = 0; o < nout; ++o) {
    float v = ob[o];
#pragma unroll
    for (int c = 0; c < 8; ++c) v = fmaf(ow[o * 8 + c], h[c], v);
    helm[((size_t)b * 4 + chbase + o) * S0C + ov] = v * sfac;
  }
}

// ======================= velocities from helmholtz =========================
__global__ void velocity_kernel(const float* __restrict__ helm, float* __restrict__ vel,
                                float* __restrict__ velp, float* __restrict__ velv) {
  int t = blockIdx.x * blockDim.x + threadIdx.x;
  if (t >= NB * TDC) return;
  int b = t >> 18, vox = t & (TDC - 1);
  int x = vox >> 12, y = (vox >> 6) & 63, z = vox & 63;
  const float* phi = helm + (size_t)(b * 4 + 0) * S0C;
  const float* s0  = helm + (size_t)(b * 4 + 1) * S0C;
  const float* s1  = helm + (size_t)(b * 4 + 2) * S0C;
  const float* s2  = helm + (size_t)(b * 4 + 3) * S0C;
  auto I = [](int X, int Y, int Z) { return (X * S0 + Y) * S0 + Z; };
  int X = x + 1, Y = y + 1, Z = z + 1;
  float pu = 0.5f * (phi[I(X, Y, z + 2)] - phi[I(X, Y, z)]);
  float pv = 0.5f * (phi[I(X, y + 2, Z)] - phi[I(X, y, Z)]);
  float pw = 0.5f * (phi[I(x + 2, Y, Z)] - phi[I(x, Y, Z)]);
  auto U = [&](int a, int bb2, int c) {
    return (s1[I(a + 1, bb2, c)] - s1[I(a, bb2, c)]) - (s0[I(a, bb2 + 1, c)] - s0[I(a, bb2, c)]);
  };
  auto V = [&](int a, int bb2, int c) {
    return (s0[I(a, bb2, c + 1)] - s0[I(a, bb2, c)]) - (s2[I(a + 1, bb2, c)] - s2[I(a, bb2, c)]);
  };
  auto W = [&](int a, int bb2, int c) {
    return (s2[I(a, bb2 + 1, c)] - s2[I(a, bb2, c)]) - (s1[I(a, bb2, c + 1)] - s1[I(a, bb2, c)]);
  };
  float vu = 0.5f * (U(X, Y, Z) + U(X, Y, z));
  float vv = 0.5f * (V(X, Y, Z) + V(X, y, Z));
  float vw = 0.5f * (W(X, Y, Z) + W(x, Y, Z));
  size_t o0 = ((size_t)b * 3 + 0) * TDC + vox;
  size_t o1 = ((size_t)b * 3 + 1) * TDC + vox;
  size_t o2 = ((size_t)b * 3 + 2) * TDC + vox;
  velp[o0] = pu; velp[o1] = pv; velp[o2] = pw;
  velv[o0] = vu; velv[o1] = vv; velv[o2] = vw;
  vel[o0] = pu + vu; vel[o1] = pv + vv; vel[o2] = pw + vw;
}

// ======================= semi-Lagrangian advection pass ====================
// mode 0: sample src0; mode 1: sample 1.5*src0 - 0.5*src1 (BFECC final pass)
__global__ void advect_kernel(const float* __restrict__ src0, const float* __restrict__ src1,
                              const float* __restrict__ vel, float sign, int mode,
                              float* __restrict__ dst) {
  int t = blockIdx.x * blockDim.x + threadIdx.x;
  if (t >= NB * TDC) return;
  int b = t >> 18, vox = t & (TDC - 1);
  int x = vox >> 12, y = (vox >> 6) & 63, z = vox & 63;
  const float* vb = vel + (size_t)b * 3 * TDC + vox;
  float vz = vb[0];
  float vy = vb[(size_t)TDC];
  float vx = vb[(size_t)2 * TDC];
  float px = fminf(fmaxf((float)x - sign * vx, 0.f), 63.f);
  float py = fminf(fmaxf((float)y - sign * vy, 0.f), 63.f);
  float pz = fminf(fmaxf((float)z - sign * vz, 0.f), 63.f);
  int x0 = (int)px, y0 = (int)py, z0 = (int)pz;
  int x1 = min(x0 + 1, 63), y1 = min(y0 + 1, 63), z1 = min(z0 + 1, 63);
  float fx = px - x0, fy = py - y0, fz = pz - z0;
  int i000 = (x0 * 64 + y0) * 64 + z0, i001 = (x0 * 64 + y0) * 64 + z1;
  int i010 = (x0 * 64 + y1) * 64 + z0, i011 = (x0 * 64 + y1) * 64 + z1;
  int i100 = (x1 * 64 + y0) * 64 + z0, i101 = (x1 * 64 + y0) * 64 + z1;
  int i110 = (x1 * 64 + y1) * 64 + z0, i111 = (x1 * 64 + y1) * 64 + z1;
  float w000 = (1 - fx) * (1 - fy) * (1 - fz), w001 = (1 - fx) * (1 - fy) * fz;
  float w010 = (1 - fx) * fy * (1 - fz),       w011 = (1 - fx) * fy * fz;
  float w100 = fx * (1 - fy) * (1 - fz),       w101 = fx * (1 - fy) * fz;
  float w110 = fx * fy * (1 - fz),             w111 = fx * fy * fz;
  for (int c = 0; c < 8; ++c) {
    const float* sp = src0 + ((size_t)b * 8 + c) * TDC;
    const float* tp = mode ? (src1 + ((size_t)b * 8 + c) * TDC) : sp;
    auto G = [&](int idx) {
      float v = sp[idx];
      return mode ? (1.5f * v - 0.5f * tp[idx]) : v;
    };
    float val = w000 * G(i000) + w001 * G(i001) + w010 * G(i010) + w011 * G(i011) +
                w100 * G(i100) + w101 * G(i101) + w110 * G(i110) + w111 * G(i111);
    dst[((size_t)b * 8 + c) * TDC + vox] = val;
  }
}

// ============ LayerNorm + FFN (64->64 GELU 64->64) + residual, WMMA ========
// pred layout (64 ch, 262144 vox). One wave handles 16 voxels x 64 features.
// A-frag (16-bit 16x32): lane element e -> k = 16*(e>>3) + 8*half + (e&7) (+32 for frag1)
// B-frag (16-bit 32x16): lane element e -> k = 16*(e>>3) + 8*half + (e&7), col = lane&15 (+32 for frag1)
// C/D 16x16 f32: vgpr r -> M = r + 8*half, N = lane&15
__global__ void __launch_bounds__(256) ffn_kernel(
    const float* __restrict__ pred,
    const float* __restrict__ lng, const float* __restrict__ lnb,
    const float* __restrict__ w1, const float* __restrict__ b1,
    const float* __restrict__ w2, const float* __restrict__ b2,
    float* __restrict__ xout) {
  __shared__ _Float16 W1h[64 * 64];
  __shared__ _Float16 W2h[64 * 64];
  __shared__ float bias1[64], bias2[64], gg[64], bb[64];
  __shared__ float hbuf[8][16 * 64];
  int tid = threadIdx.x;
  for (int i = tid; i < 4096; i += 256) {
    W1h[i] = (_Float16)w1[i];
    W2h[i] = (_Float16)w2[i];
  }
  if (tid < 64) { bias1[tid] = b1[tid]; bias2[tid] = b2[tid]; gg[tid] = lng[tid]; bb[tid] = lnb[tid]; }
  __syncthreads();
  int wv = tid >> 5, lane = tid & 31;
  int m = lane & 15, half = lane >> 4;
  int vbase = (blockIdx.x * 8 + wv) * 16;
  int v = vbase + m;

  // load this lane's 32 of the 64 channels for row m (A-frag layout), LN-reduce with partner lane
  float av[32];
#pragma unroll
  for (int i = 0; i < 16; ++i) {
    int k = ((i >> 3) << 4) + (half << 3) + (i & 7);
    av[i]      = pred[(size_t)k * TDC + v];
    av[16 + i] = pred[(size_t)(32 + k) * TDC + v];
  }
  float s = 0.f, q = 0.f;
#pragma unroll
  for (int i = 0; i < 32; ++i) { s += av[i]; q += av[i] * av[i]; }
  s += __shfl_xor(s, 16, 32);
  q += __shfl_xor(q, 16, 32);
  float mean = s * (1.0f / 64.0f);
  float var  = q * (1.0f / 64.0f) - mean * mean;
  float rstd = rsqrtf(var + EPSV);
  v16h a0, a1;
#pragma unroll
  for (int i = 0; i < 16; ++i) {
    int k = ((i >> 3) << 4) + (half << 3) + (i & 7);
    a0[i] = (_Float16)((av[i] - mean) * rstd * gg[k] + bb[k]);
    a1[i] = (_Float16)((av[16 + i] - mean) * rstd * gg[32 + k] + bb[32 + k]);
  }

  // GEMM1: h1 = gelu(a @ W1 + b1) -> hbuf
#pragma unroll
  for (int nt = 0; nt < 4; ++nt) {
    int col = nt * 16 + (lane & 15);
    v16h bf0, bf1;
#pragma unroll
    for (int i = 0; i < 16; ++i) {
      int k0 = ((i >> 3) << 4) + (half << 3) + (i & 7);
      bf0[i] = W1h[k0 * 64 + col];
      bf1[i] = W1h[(32 + k0) * 64 + col];
    }
    v8f acc;
    float bv = bias1[col];
#pragma unroll
    for (int r = 0; r < 8; ++r) acc[r] = bv;
    acc = __builtin_amdgcn_wmma_f32_16x16x32_f16(false, a0, false, bf0, (short)0, acc, false, false);
    acc = __builtin_amdgcn_wmma_f32_16x16x32_f16(false, a1, false, bf1, (short)0, acc, false, false);
#pragma unroll
    for (int r = 0; r < 8; ++r) {
      float xv = acc[r];
      float gl = 0.5f * xv * (1.0f + erff(xv * 0.70710678118f));
      int M = r + 8 * half;
      hbuf[wv][M * 64 + col] = gl;
    }
  }
  __syncthreads();

  // reload h1 as A-frags
  v16h c0, c1;
#pragma unroll
  for (int i = 0; i < 16; ++i) {
    int k = ((i >> 3) << 4) + (half << 3) + (i & 7);
    c0[i] = (_Float16)hbuf[wv][m * 64 + k];
    c1[i] = (_Float16)hbuf[wv][m * 64 + 32 + k];
  }

  // GEMM2 + residual
#pragma unroll
  for (int nt = 0; nt < 4; ++nt) {
    int col = nt * 16 + (lane & 15);
    v16h df0, df1;
#pragma unroll
    for (int i = 0; i < 16; ++i) {
      int k0 = ((i >> 3) << 4) + (half << 3) + (i & 7);
      df0[i] = W2h[k0 * 64 + col];
      df1[i] = W2h[(32 + k0) * 64 + col];
    }
    v8f acc;
    float bv = bias2[col];
#pragma unroll
    for (int r = 0; r < 8; ++r) acc[r] = bv;
    acc = __builtin_amdgcn_wmma_f32_16x16x32_f16(false, c0, false, df0, (short)0, acc, false, false);
    acc = __builtin_amdgcn_wmma_f32_16x16x32_f16(false, c1, false, df1, (short)0, acc, false, false);
#pragma unroll
    for (int r = 0; r < 8; ++r) {
      int M = r + 8 * half;
      int vv = vbase + M;
      xout[(size_t)col * TDC + vv] = acc[r] + pred[(size_t)col * TDC + vv];
    }
  }
}

// ===========================================================================
extern "C" void kernel_launch(void* const* d_in, const int* in_sizes, int n_in,
                              void* d_out_v, int out_size, void* d_ws, size_t ws_size,
                              hipStream_t stream) {
  (void)in_sizes; (void)n_in; (void)out_size; (void)ws_size;
  const float* prev       = (const float*)d_in[0];
  const float* nxt        = (const float*)d_in[1];
  const float* texture    = (const float*)d_in[2];
  const float* mask       = (const float*)d_in[3];
  const float* boundary   = (const float*)d_in[4];
  const float* enc_w      = (const float*)d_in[5];
  const float* enc_b      = (const float*)d_in[6];
  const float* phi_c1_w   = (const float*)d_in[7];
  const float* phi_bn1_g  = (const float*)d_in[8];
  const float* phi_bn1_b  = (const float*)d_in[9];
  const float* phi_c2_w   = (const float*)d_in[10];
  const float* phi_bn2_g  = (const float*)d_in[11];
  const float* phi_bn2_b  = (const float*)d_in[12];
  const float* phi_out_w  = (const float*)d_in[13];
  const float* phi_out_b  = (const float*)d_in[14];
  const float* vort_c1_w  = (const float*)d_in[15];
  const float* vort_bn1_g = (const float*)d_in[16];
  const float* vort_bn1_b = (const float*)d_in[17];
  const float* vort_c2_w  = (const float*)d_in[18];
  const float* vort_bn2_g = (const float*)d_in[19];
  const float* vort_bn2_b = (const float*)d_in[20];
  const float* vort_out_w = (const float*)d_in[21];
  const float* vort_out_b = (const float*)d_in[22];
  const float* phi_weight = (const float*)d_in[23];
  const float* vort_weight= (const float*)d_in[24];
  const float* ln_g       = (const float*)d_in[25];
  const float* ln_b       = (const float*)d_in[26];
  const float* ff_w1      = (const float*)d_in[27];
  const float* ff_b1      = (const float*)d_in[28];
  const float* ff_w2      = (const float*)d_in[29];
  const float* ff_b2      = (const float*)d_in[30];

  float* d_out = (float*)d_out_v;
  char* ws = (char*)d_ws;
  const size_t MiB = 1ull << 20;
  // lifetime-overlapped arena; peak requirement ~221 MiB
  float* CORR  = (float*)(ws + 0);
  float* F0    = (float*)(ws + 64 * MiB);
  float* F1    = (float*)(ws + 144 * MiB);
  float* C1P   = (float*)(ws + 200 * MiB);
  float* C1V   = (float*)(ws + 210 * MiB);
  float* STATS = (float*)(ws + 220 * MiB);
  float* PART  = (float*)(ws + 220 * MiB + 1024);
  float* UP    = (float*)(ws + 0);
  float* C2    = (float*)(ws + 80 * MiB);
  float* PHI1  = (float*)(ws + 0);
  float* PHI2  = (float*)(ws + 80 * MiB);
  float* PRED  = (float*)(ws + 152 * MiB);

  int n1 = NB * S1C, g1 = (n1 + 255) / 256;   // 33^3 grids
  int n2 = NB * S0C, g2 = (n2 + 255) / 256;   // 66^3 grids
  int n3 = NB * TDC, g3 = n3 / 256;           // 64^3 grids
  (void)n2; (void)n3;

  // encoder + correlation (f_pm/f_pb sequenced through F1 to cut ws peak)
  enc_kernel<<<g1, 256, 0, stream>>>(nxt,  mask,     enc_w, enc_b, F0);
  enc_kernel<<<g1, 256, 0, stream>>>(prev, mask,     enc_w, enc_b, F1);
  corr_kernel<<<g1, 256, 0, stream>>>(F1, F0, CORR, 0);
  enc_kernel<<<g1, 256, 0, stream>>>(prev, boundary, enc_w, enc_b, F1);
  corr_kernel<<<g1, 256, 0, stream>>>(F1, F0, CORR, 27);

  // conv1 for both branches (CORR dead afterwards)
  conv1_kernel<<<g1, 256, 0, stream>>>(CORR, phi_c1_w,  C1P);
  conv1_kernel<<<g1, 256, 0, stream>>>(CORR, vort_c1_w, C1V);

  struct Br {
    const float *g1p, *b1p, *c2w, *g2p, *b2p, *ow, *ob, *wsc;
    float wmul; int chbase; int nout; float* c1;
  };
  Br brs[2] = {
    { phi_bn1_g,  phi_bn1_b,  phi_c2_w,  phi_bn2_g,  phi_bn2_b,  phi_out_w,  phi_out_b,  phi_weight,  1.0f,  0, 1, C1P },
    { vort_bn1_g, vort_bn1_b, vort_c2_w, vort_bn2_g, vort_bn2_b, vort_out_w, vort_out_b, vort_weight, 66.0f, 1, 3, C1V },
  };
  for (int i = 0; i < 2; ++i) {
    Br& B = brs[i];
    bnstat1_kernel<<<dim3(64, 8), 256, 0, stream>>>(B.c1, PART, S1C, 8, NB);
    bnstat2_kernel<<<8, 64, 0, stream>>>(PART, STATS, 64);
    upsample_kernel<<<g2, 256, 0, stream>>>(B.c1, STATS, B.g1p, B.b1p, 1.0f / (float)(NB * S1C), UP);
    conv2_kernel<<<g2, 256, 0, stream>>>(UP, B.c2w, C2);
    bnstat1_kernel<<<dim3(64, 8), 256, 0, stream>>>(C2, PART, S0C, 8, NB);
    bnstat2_kernel<<<8, 64, 0, stream>>>(PART, STATS, 64);
    finalize_kernel<<<g2, 256, 0, stream>>>(C2, STATS, B.g2p, B.b2p, B.ow, B.ob, B.wsc, B.wmul,
                                            1.0f / (float)(NB * S0C),
                                            d_out + HELM_OFF, B.chbase, B.nout);
  }

  // velocities (reads helmholtz from d_out, writes vel/vel_phi/vel_vort)
  velocity_kernel<<<g3, 256, 0, stream>>>(d_out + HELM_OFF, d_out + VEL_OFF,
                                          d_out + VELP_OFF, d_out + VELV_OFF);

  // BFECC: phi1 = A(tex, +v); phi2 = A(phi1, -v); pred = A(1.5 tex - 0.5 phi2, +v)
  advect_kernel<<<g3, 256, 0, stream>>>(texture, texture, d_out + VEL_OFF,  1.0f, 0, PHI1);
  advect_kernel<<<g3, 256, 0, stream>>>(PHI1,    PHI1,    d_out + VEL_OFF, -1.0f, 0, PHI2);
  advect_kernel<<<g3, 256, 0, stream>>>(texture, PHI2,    d_out + VEL_OFF,  1.0f, 1, PRED);

  // LayerNorm + FFN (WMMA) + residual -> x
  ffn_kernel<<<TDC / 128, 256, 0, stream>>>(PRED, ln_g, ln_b, ff_w1, ff_b1, ff_w2, ff_b2,
                                            d_out + X_OFF);
}
